// NonstationaryGaussianSpectralMixtureKernel_19224273617256
// MI455X (gfx1250) — matrix operand order repaired and checked by reading, not compiled, not run on hardware
//
#include <hip/hip_runtime.h>

typedef __attribute__((ext_vector_type(16))) _Float16 v16h;
typedef __attribute__((ext_vector_type(8)))  float    v8f;

#define Q      8
#define LATENT 64
#define OUT_F  32
#define NPTS   2048
#define LOG2E  1.4426950408889634f
#define LN2    0.6931471805599453f

__device__ __forceinline__ float fast_exp(float v) {
  return __builtin_amdgcn_exp2f(v * LOG2E);
}
__device__ __forceinline__ float softplus_f(float v) {
  float ax = fabsf(v);
  float e  = __builtin_amdgcn_exp2f(-ax * LOG2E);
  // __builtin_amdgcn_logf == v_log_f32 == log2
  return fmaxf(v, 0.0f) + LN2 * __builtin_amdgcn_logf(1.0f + e);
}
__device__ __forceinline__ float selu_f(float v) {
  const float scale = 1.0507009873554805f, alpha = 1.6732632423543772f;
  float neg = scale * alpha * (fast_exp(v) - 1.0f);
  return v > 0.0f ? scale * v : neg;
}

// Phase 1: MLP features for all 4096 points (2048 x + 2048 y).
// Layer 2 (h[64] @ W2^T -> 32 outs) done with v_wmma_f32_16x16x32_f16:
// each wave handles 16 points; 2 K-chunks x 2 N-tiles = 4 WMMAs.
__global__ __launch_bounds__(128) void feat_kernel(
    const float* __restrict__ x, const float* __restrict__ y,
    const float* __restrict__ W1, const float* __restrict__ b1,
    const float* __restrict__ W2, const float* __restrict__ b2,
    float4* __restrict__ featA, float4* __restrict__ featB) {
  __shared__ _Float16 h_lds[64 * LATENT];   // 8 KB
  __shared__ float    a_lds[64 * OUT_F];    // 8 KB
  const int tid = threadIdx.x;
  const int blockBase = blockIdx.x * 64;

  // ---- layer 1: h = selu(p @ W1^T + b1), 64 points x 64 hidden ----
  #pragma unroll 4
  for (int t = 0; t < 32; ++t) {
    int idx = t * 128 + tid;
    int pt = idx >> 6, hid = idx & 63;
    int pl = blockBase + pt;
    const float* P = (pl < NPTS) ? (x + 2 * pl) : (y + 2 * (pl - NPTS));
    float v = fmaf(P[0], W1[2 * hid], fmaf(P[1], W1[2 * hid + 1], b1[hid]));
    h_lds[pt * LATENT + hid] = (_Float16)selu_f(v);
  }
  __syncthreads();

  // ---- layer 2 via WMMA ----
  const int lane = tid & 31;
  const int wave = tid >> 5;
  const int wpt  = wave * 16;       // local point base of this wave
  const int row  = lane & 15;       // A/D row, B/D col index
  const int hi   = lane >> 4;       // lane-half selector

  // A fragments: 16x32 f16 tiles, K-chunk 0 (K=0..31) and 1 (K=32..63)
  v16h a0, a1;
  #pragma unroll
  for (int e = 0; e < 16; ++e) {
    int kl = (e < 8) ? (hi * 8 + e) : (16 + hi * 8 + (e - 8));
    a0[e] = h_lds[(wpt + row) * LATENT + kl];
    a1[e] = h_lds[(wpt + row) * LATENT + 32 + kl];
  }
  #pragma unroll
  for (int n0 = 0; n0 < OUT_F; n0 += 16) {
    const int n = n0 + row;         // output-neuron column
    v16h bf0, bf1;                  // B[k][n] = W2[n][k]
    #pragma unroll
    for (int e = 0; e < 16; ++e) {
      int k = (e < 8) ? (hi * 8 + e) : (16 + hi * 8 + (e - 8));
      bf0[e] = (_Float16)W2[n * LATENT + k];
      bf1[e] = (_Float16)W2[n * LATENT + 32 + k];
    }
    v8f c;
    float bias = b2[n];
    #pragma unroll
    for (int r = 0; r < 8; ++r) c[r] = bias;
    c = __builtin_amdgcn_wmma_f32_16x16x32_f16(false, a0, false, bf0,
                                               (short)0, c, false, false);
    c = __builtin_amdgcn_wmma_f32_16x16x32_f16(false, a1, false, bf1,
                                               (short)0, c, false, false);
    #pragma unroll
    for (int r = 0; r < 8; ++r) {
      int m = r + hi * 8;           // C/D: lanes 0-15 -> M=r, lanes 16-31 -> M=8+r
      a_lds[(wpt + m) * OUT_F + n] = softplus_f(c[r]);
    }
  }
  __syncthreads();

  // ---- derived features: (u, s^2, cos 2pi*phi, sin 2pi*phi) ----
  #pragma unroll
  for (int t = 0; t < 4; ++t) {
    int idx = t * 128 + tid;
    int pt = idx >> 3, q = idx & 7;
    int pl = blockBase + pt;
    const float* P = (pl < NPTS) ? (x + 2 * pl) : (y + 2 * (pl - NPTS));
    const float* arow = a_lds + pt * OUT_F;
    float w  = arow[q];
    float s  = arow[8 + q];
    float f0 = arow[16 + 2 * q];
    float f1 = arow[17 + 2 * q];
    float phi = fmaf(f0, P[0], f1 * P[1]);
    float4 ft;
    ft.x = w * __builtin_amdgcn_sqrtf(s) * 1.1892071150027210f; // w*sqrt(s)*2^0.25
    ft.y = s * s;
    ft.z = __builtin_amdgcn_cosf(phi);  // v_cos_f32: input in revolutions -> cos(2*pi*phi)
    ft.w = __builtin_amdgcn_sinf(phi);
    float4* dst = (pl < NPTS) ? (featA + pl * Q + q) : (featB + (pl - NPTS) * Q + q);
    *dst = ft;
  }
}

// Phase 2: 2048x2048 pairwise kernel, 64x64 tile per block,
// 4x4 register micro-tile per thread, features staged in LDS.
__global__ __launch_bounds__(256) void pair_kernel(
    const float* __restrict__ x, const float* __restrict__ y,
    const float4* __restrict__ featA, const float4* __restrict__ featB,
    float* __restrict__ out) {
  __shared__ float4 Af[64 * Q];   // 8 KB
  __shared__ float4 Bf[64 * Q];   // 8 KB
  __shared__ float2 axy[64];
  __shared__ float2 bxy[64];
  const int tid = threadIdx.x;
  const int i0 = blockIdx.y * 64;
  const int j0 = blockIdx.x * 64;

  Af[tid]       = featA[i0 * Q + tid];
  Af[tid + 256] = featA[i0 * Q + tid + 256];
  Bf[tid]       = featB[j0 * Q + tid];
  Bf[tid + 256] = featB[j0 * Q + tid + 256];
  if (tid < 64) {
    axy[tid] = ((const float2*)x)[i0 + tid];
    bxy[tid] = ((const float2*)y)[j0 + tid];
  }
  __syncthreads();

  const int tx = tid & 15, ty = tid >> 4;
  float2 ai[4], bj[4];
  #pragma unroll
  for (int ii = 0; ii < 4; ++ii) ai[ii] = axy[ty * 4 + ii];
  #pragma unroll
  for (int jj = 0; jj < 4; ++jj) bj[jj] = bxy[tx * 4 + jj];

  float d2[4][4], acc[4][4];
  #pragma unroll
  for (int ii = 0; ii < 4; ++ii)
    #pragma unroll
    for (int jj = 0; jj < 4; ++jj) {
      float dx = ai[ii].x - bj[jj].x;
      float dy = ai[ii].y - bj[jj].y;
      d2[ii][jj]  = fmaf(dx, dx, dy * dy);
      acc[ii][jj] = 0.0f;
    }

  #pragma unroll
  for (int q = 0; q < Q; ++q) {
    float4 A[4], B[4];
    #pragma unroll
    for (int ii = 0; ii < 4; ++ii) A[ii] = Af[(ty * 4 + ii) * Q + q];
    #pragma unroll
    for (int jj = 0; jj < 4; ++jj) B[jj] = Bf[(tx * 4 + jj) * Q + q];
    #pragma unroll
    for (int ii = 0; ii < 4; ++ii)
      #pragma unroll
      for (int jj = 0; jj < 4; ++jj) {
        float s2  = A[ii].y + B[jj].y;                       // sA^2 + sB^2
        float inv = __builtin_amdgcn_rcpf(s2);
        float e   = __builtin_amdgcn_exp2f(-d2[ii][jj] * inv * LOG2E);
        float cs  = fmaf(A[ii].z, B[jj].z, A[ii].w * B[jj].w); // cos(2pi(phiB-phiA))
        acc[ii][jj] = fmaf(A[ii].x * B[jj].x, inv * e * cs, acc[ii][jj]);
      }
  }

  #pragma unroll
  for (int ii = 0; ii < 4; ++ii) {
    float4 v = make_float4(acc[ii][0], acc[ii][1], acc[ii][2], acc[ii][3]);
    float4* orow = (float4*)(out + (size_t)(i0 + ty * 4 + ii) * 2048 + j0);
    orow[tx] = v;   // coalesced 16B stores, full row segment
  }
}

extern "C" void kernel_launch(void* const* d_in, const int* in_sizes, int n_in,
                              void* d_out, int out_size, void* d_ws, size_t ws_size,
                              hipStream_t stream) {
  const float* x  = (const float*)d_in[0];
  const float* y  = (const float*)d_in[1];
  const float* W1 = (const float*)d_in[2];
  const float* b1 = (const float*)d_in[3];
  const float* W2 = (const float*)d_in[4];
  const float* b2 = (const float*)d_in[5];
  float* out = (float*)d_out;

  float4* featA = (float4*)d_ws;               // 2048*8 float4 = 256 KB
  float4* featB = featA + (size_t)NPTS * Q;    // 256 KB

  feat_kernel<<<64, 128, 0, stream>>>(x, y, W1, b1, W2, b2, featA, featB);
  pair_kernel<<<dim3(32, 32), 256, 0, stream>>>(x, y, featA, featB, out);
}
